// SAGNetwork_85890755985941
// MI455X (gfx1250) — matrix operand order repaired and checked by reading, not compiled, not run on hardware
//
#include <hip/hip_runtime.h>
#include <hip/hip_bf16.h>
#include <math.h>

// Problem constants (match reference)
#define BATCH 1024
#define NNODE 64
#define OBSD  128
#define HID   256
#define ACTD  16
#define NHEAD 4
#define HDIM  64

typedef __bf16 bf16;
typedef __attribute__((ext_vector_type(16))) __bf16 v16bf;
typedef __attribute__((ext_vector_type(8)))  __bf16 v8bf;
typedef __attribute__((ext_vector_type(8)))  float  v8f;

// LDS (addrspace 3) typed pointers: all LDS arithmetic stays 32-bit and
// generates ds_* directly with no generic-pointer null guards.
#define LDS_AS __attribute__((address_space(3)))
typedef LDS_AS bf16 lbf16;
typedef LDS_AS v8bf lv8bf;

// LDS pitches (in bf16 elements); every pitch*2 bytes is a multiple of 16B
#define PX   264   // 256-wide activation buffers
#define POBS 136   // 128-wide obs buffer
#define PVT  72    // Vt / scores / adj pitch (64-wide)

// LDS byte offsets (all 16B aligned)
#define OFF_S    0                              // S [4][64][72] bf16 (aliases obs staging)
#define OFF_XA   36864                          // XA [64][264] (x1, then K-projection)
#define OFF_XB   (36864 + 33792)                // XB [64][264] (x2, xo, z)
#define OFF_Q    (36864 + 2*33792)              // Q  [64][264] (q, then attn-out O, then c)
#define OFF_VT   (36864 + 3*33792)              // Vt [256][72] (V transposed: feature-major)
#define OFF_ADJ  (36864 + 3*33792 + 36864)      // ADJ [64][72] bf16
#define SMEM_BYTES (36864 + 3*33792 + 36864 + 9216)   // 184320 B < 320 KB WGP LDS

// ---------------------------------------------------------------------------
// WMMA wrapper: D = A(16x32 bf16) * B(32x16 bf16) + C(16x16 f32)
// ---------------------------------------------------------------------------
__device__ __forceinline__ v8f wmma_bf16(v16bf a, v16bf b, v8f c) {
  return __builtin_amdgcn_wmma_f32_16x16x32_bf16(
      /*neg_a=*/false, a, /*neg_b=*/false, b,
      /*c_mod=*/(short)0, c, /*reuse_a=*/false, /*reuse_b=*/false);
}

// A-fragment (16x32) from row-major bf16 LDS. Contiguous k order used here:
// lane r(=lane&15) holds row M=r; halfs i -> K=g*16+i (matches B fragments).
// Two contiguous 16B chunks per lane -> ds_load_b128 x2.
__device__ __forceinline__ v16bf load_a(const lbf16* base, int pitch, int row0, int k0) {
  const int lane = threadIdx.x & 31, r = lane & 15, g = lane >> 4;
  const lbf16* p = base + (row0 + r) * pitch + k0 + g * 16;
  v8bf lo = *(const lv8bf*)(p);
  v8bf hi = *(const lv8bf*)(p + 8);
  v16bf out;
#pragma unroll
  for (int i = 0; i < 8; ++i) { out[i] = lo[i]; out[i + 8] = hi[i]; }
  return out;
}

// B-fragment (32x16) where LDS storage is row-major by COLUMN index: storage[c][k]
// (K-matrix rows for Q.K^T, transposed V). Contiguous k order used here:
// lane r -> column N=r; halfs i -> K = g*16 + i (one contiguous 32B run).
__device__ __forceinline__ v16bf load_b_rows(const lbf16* base, int pitch, int c0, int k0) {
  const int lane = threadIdx.x & 31, r = lane & 15, g = lane >> 4;
  const lbf16* p = base + (c0 + r) * pitch + k0 + g * 16;
  v8bf lo = *(const lv8bf*)(p);
  v8bf hi = *(const lv8bf*)(p + 8);
  v16bf out;
#pragma unroll
  for (int i = 0; i < 8; ++i) { out[i] = lo[i]; out[i + 8] = hi[i]; }
  return out;
}

// B-fragment from pre-packed GLOBAL weights: block = 32 lanes x 16 halfs
// contiguous -> two global_load_b128 (uniform SGPR base + small int offset).
__device__ __forceinline__ v16bf load_b_packed(const bf16* __restrict__ Wp, int blkoff) {
  const int lane = threadIdx.x & 31;
  const v8bf* p = (const v8bf*)(Wp + blkoff + lane * 16);
  v8bf lo = p[0];
  v8bf hi = p[1];
  v16bf out;
#pragma unroll
  for (int i = 0; i < 8; ++i) { out[i] = lo[i]; out[i + 8] = hi[i]; }
  return out;
}

// B-fragment from K-major LDS storage (storage[k][c]) -> 16 strided u16 loads.
__device__ __forceinline__ v16bf load_b_strided(const lbf16* base, int pitch, int k0, int c0) {
  const int lane = threadIdx.x & 31, c = c0 + (lane & 15), g = lane >> 4;
  v16bf out;
#pragma unroll
  for (int i = 0; i < 16; ++i)
    out[i] = base[(k0 + g * 16 + i) * pitch + c];
  return out;
}

// C/D layout: VGPR j holds (M = g*8+j, N = lane&15).
__device__ __forceinline__ void store_c(lbf16* base, int pitch, int row0, int col0,
                                        v8f c, bool relu) {
  const int lane = threadIdx.x & 31, r = lane & 15, g = lane >> 4;
#pragma unroll
  for (int j = 0; j < 8; ++j) {
    float v = c[j];
    if (relu) v = v > 0.f ? v : 0.f;
    base[(row0 + g * 8 + j) * pitch + col0 + r] = (bf16)v;
  }
}
__device__ __forceinline__ void store_c_T(lbf16* base, int pitch, int row0, int col0, v8f c) {
  const int lane = threadIdx.x & 31, r = lane & 15, g = lane >> 4;
#pragma unroll
  for (int j = 0; j < 8; ++j)
    base[(col0 + r) * pitch + row0 + g * 8 + j] = (bf16)c[j];
}

// ---------------------------------------------------------------------------
// Dense GEMM: Y[64,256] = act( X[64,32*KT] * W^T + b ).  8 waves; wave owns a
// 32(M) x 64(N) stripe = 2x4 WMMA tiles.  KT is a compile-time constant so all
// weight-block offsets are immediates on the uniform SGPR base pointer.
// ---------------------------------------------------------------------------
template <int KT, bool RELU, bool TRANSPOSED>
__device__ __forceinline__ void gemm_dense(const lbf16* X, int pitchX,
                                           const bf16* __restrict__ Wp,
                                           const float* __restrict__ bias,
                                           lbf16* Y, int pitchY) {
  const int wave = threadIdx.x >> 5;
  const int lane = threadIdx.x & 31, r = lane & 15;
  const int m0 = (wave >> 2) * 32;
  const int n0 = (wave & 3) * 64;

  v8f acc[2][4];
#pragma unroll
  for (int nt = 0; nt < 4; ++nt) {
    float bv = bias ? bias[n0 + nt * 16 + r] : 0.f;
    v8f z;
#pragma unroll
    for (int j = 0; j < 8; ++j) z[j] = bv;
    acc[0][nt] = z;
    acc[1][nt] = z;
  }

#pragma unroll
  for (int kt = 0; kt < KT; ++kt) {
    v16bf a0 = load_a(X, pitchX, m0,      kt * 32);
    v16bf a1 = load_a(X, pitchX, m0 + 16, kt * 32);
    if (kt + 1 < KT)  // global_prefetch_b8 of next K-block of our first N tile
      __builtin_prefetch(Wp + ((n0 >> 4) * KT + kt + 1) * 512 + lane * 16, 0, 1);
#pragma unroll
    for (int nt = 0; nt < 4; ++nt) {
      v16bf bw = load_b_packed(Wp, (((n0 >> 4) + nt) * KT + kt) * 512);
      acc[0][nt] = wmma_bf16(a0, bw, acc[0][nt]);
      acc[1][nt] = wmma_bf16(a1, bw, acc[1][nt]);
    }
  }

#pragma unroll
  for (int mt = 0; mt < 2; ++mt)
#pragma unroll
    for (int nt = 0; nt < 4; ++nt) {
      if (TRANSPOSED) store_c_T(Y, pitchY, m0 + mt * 16, n0 + nt * 16, acc[mt][nt]);
      else            store_c  (Y, pitchY, m0 + mt * 16, n0 + nt * 16, acc[mt][nt], RELU);
    }
}

// ---------------------------------------------------------------------------
// Fused SAG network: one workgroup (256 threads / 8 waves) per batch graph.
// ---------------------------------------------------------------------------
__global__ __launch_bounds__(256, 1)
void sag_fused(const float* __restrict__ obs, const float* __restrict__ adj,
               const bf16* __restrict__ W1p, const float* __restrict__ b1,
               const bf16* __restrict__ W2p, const float* __restrict__ b2,
               const bf16* __restrict__ Wqp, const float* __restrict__ bq,
               const bf16* __restrict__ Wkp, const float* __restrict__ bk,
               const bf16* __restrict__ Wvp, const float* __restrict__ bv,
               const bf16* __restrict__ Wop, const float* __restrict__ bo,
               const bf16* __restrict__ Wcp, const float* __restrict__ bc,
               const bf16* __restrict__ W3p, const float* __restrict__ b3,
               float* __restrict__ out) {
  extern __shared__ __align__(16) char smem[];
  // Single generic->LDS cast; all later arithmetic stays in addrspace(3).
  LDS_AS char* sb = (LDS_AS char*)smem;
  lbf16* Sl   = (lbf16*)(sb + OFF_S);     // scores (later); obs staging (now)
  lbf16* OBSl = Sl;                       // alias
  lbf16* XAl  = (lbf16*)(sb + OFF_XA);
  lbf16* XBl  = (lbf16*)(sb + OFF_XB);
  lbf16* Qb   = (lbf16*)(sb + OFF_Q);
  lbf16* Vt   = (lbf16*)(sb + OFF_VT);
  lbf16* ADJl = (lbf16*)(sb + OFF_ADJ);

  const int b    = blockIdx.x;
  const int wave = threadIdx.x >> 5;
  const int lane = threadIdx.x & 31, r = lane & 15, g = lane >> 4;
  const int gm0  = (wave >> 2) * 32;
  const int gn0  = (wave & 3) * 64;

  // ---- stage obs & adj into LDS as bf16 -------------------------------------
  {
    const float* obsb = obs + (size_t)b * NNODE * OBSD;
    for (int idx = threadIdx.x; idx < NNODE * OBSD; idx += 256) {
      int rr = idx >> 7, cc = idx & 127;
      OBSl[rr * POBS + cc] = (bf16)obsb[idx];
    }
    const float* adjb = adj + (size_t)b * NNODE * NNODE;
    for (int idx = threadIdx.x; idx < NNODE * NNODE; idx += 256) {
      int rr = idx >> 6, cc = idx & 63;
      ADJl[rr * PVT + cc] = (bf16)adjb[idx];   // 0/1 exact in bf16
    }
  }
  __syncthreads();

  // ---- x1 = relu(obs @ W1^T + b1) ------------------------------------------
  gemm_dense<OBSD / 32, true, false>(OBSl, POBS, W1p, b1, XAl, PX);
  __syncthreads();
  // ---- x2 = relu(x1 @ W2^T + b2) -------------------------------------------
  gemm_dense<HID / 32, true, false>(XAl, PX, W2p, b2, XBl, PX);
  __syncthreads();
  // ---- q, k, v projections (v stored transposed: Vt[feature][node]) --------
  gemm_dense<HID / 32, false, false>(XBl, PX, Wqp, bq, Qb, PX);
  gemm_dense<HID / 32, false, false>(XBl, PX, Wkp, bk, XAl, PX);   // K reuses XA
  gemm_dense<HID / 32, false, true >(XBl, PX, Wvp, bv, Vt, PVT);
  __syncthreads();

  // ---- attention scores + masked softmax: 2 waves per head ------------------
  {
    const int h = wave >> 1;
    const int m0 = (wave & 1) * 32;
    lbf16* Sh = Sl + h * NNODE * PVT;

    v8f s[2][4];
#pragma unroll
    for (int mt = 0; mt < 2; ++mt)
#pragma unroll
      for (int nt = 0; nt < 4; ++nt) {
        v8f z;
#pragma unroll
        for (int j = 0; j < 8; ++j) z[j] = 0.f;
        s[mt][nt] = z;
      }

#pragma unroll
    for (int ks = 0; ks < 2; ++ks) {
      const int k0 = h * HDIM + ks * 32;
      v16bf a0 = load_a(Qb, PX, m0,      k0);
      v16bf a1 = load_a(Qb, PX, m0 + 16, k0);
#pragma unroll
      for (int nt = 0; nt < 4; ++nt) {
        v16bf bK = load_b_rows(XAl, PX, nt * 16, k0);  // K rows == B^T
        s[0][nt] = wmma_bf16(a0, bK, s[0][nt]);
        s[1][nt] = wmma_bf16(a1, bK, s[1][nt]);
      }
    }

    const float scale = 0.125f;  // 1/sqrt(HD=64)
#pragma unroll
    for (int mt = 0; mt < 2; ++mt) {
#pragma unroll
      for (int j = 0; j < 8; ++j) {
        const int row = m0 + mt * 16 + g * 8 + j;
        float v[4];
#pragma unroll
        for (int nt = 0; nt < 4; ++nt) {
          float a = (float)ADJl[row * PVT + nt * 16 + r];
          v[nt] = (a != 0.f) ? s[mt][nt][j] * scale : -3.0e38f;
        }
        float mx = fmaxf(fmaxf(v[0], v[1]), fmaxf(v[2], v[3]));
#pragma unroll
        for (int off = 1; off < 16; off <<= 1) mx = fmaxf(mx, __shfl_xor(mx, off, 32));
        float e[4], sum = 0.f;
#pragma unroll
        for (int nt = 0; nt < 4; ++nt) { e[nt] = __expf(v[nt] - mx); sum += e[nt]; }
#pragma unroll
        for (int off = 1; off < 16; off <<= 1) sum += __shfl_xor(sum, off, 32);
        const float inv = 1.f / sum;
#pragma unroll
        for (int nt = 0; nt < 4; ++nt)
          Sh[row * PVT + nt * 16 + r] = (bf16)(e[nt] * inv);
      }
    }
  }
  __syncthreads();

  // ---- O = attn @ V  (per head, into Qb; Q is dead) -------------------------
  {
    const int h = wave >> 1;
    const int m0 = (wave & 1) * 32;
    const lbf16* Sh = Sl + h * NNODE * PVT;
    v8f o[2][4];
#pragma unroll
    for (int mt = 0; mt < 2; ++mt)
#pragma unroll
      for (int nt = 0; nt < 4; ++nt) {
        v8f z;
#pragma unroll
        for (int j = 0; j < 8; ++j) z[j] = 0.f;
        o[mt][nt] = z;
      }
#pragma unroll
    for (int ks = 0; ks < 2; ++ks) {
      v16bf a0 = load_a(Sh, PVT, m0,      ks * 32);
      v16bf a1 = load_a(Sh, PVT, m0 + 16, ks * 32);
#pragma unroll
      for (int nt = 0; nt < 4; ++nt) {
        v16bf bV = load_b_rows(Vt, PVT, h * HDIM + nt * 16, ks * 32);
        o[0][nt] = wmma_bf16(a0, bV, o[0][nt]);
        o[1][nt] = wmma_bf16(a1, bV, o[1][nt]);
      }
    }
#pragma unroll
    for (int mt = 0; mt < 2; ++mt)
#pragma unroll
      for (int nt = 0; nt < 4; ++nt)
        store_c(Qb, PX, m0 + mt * 16, h * HDIM + nt * 16, o[mt][nt], false);
  }
  __syncthreads();

  // ---- xo = O @ Wo^T + bo ---------------------------------------------------
  gemm_dense<HID / 32, false, false>(Qb, PX, Wop, bo, XBl, PX);
  __syncthreads();
  // ---- c = xo @ Wc^T + bc  (into Qb; O is dead) -----------------------------
  gemm_dense<HID / 32, false, false>(XBl, PX, Wcp, bc, Qb, PX);
  __syncthreads();

  // ---- z = adj @ c  (K = 64 node dim; B is K-major -> strided loads) --------
  {
    v8f acc[2][4];
#pragma unroll
    for (int mt = 0; mt < 2; ++mt)
#pragma unroll
      for (int nt = 0; nt < 4; ++nt) {
        v8f z;
#pragma unroll
        for (int j = 0; j < 8; ++j) z[j] = 0.f;
        acc[mt][nt] = z;
      }
#pragma unroll
    for (int kt = 0; kt < 2; ++kt) {
      v16bf a0 = load_a(ADJl, PVT, gm0,      kt * 32);
      v16bf a1 = load_a(ADJl, PVT, gm0 + 16, kt * 32);
#pragma unroll
      for (int nt = 0; nt < 4; ++nt) {
        v16bf bc_ = load_b_strided(Qb, PX, kt * 32, gn0 + nt * 16);
        acc[0][nt] = wmma_bf16(a0, bc_, acc[0][nt]);
        acc[1][nt] = wmma_bf16(a1, bc_, acc[1][nt]);
      }
    }
#pragma unroll
    for (int mt = 0; mt < 2; ++mt)
#pragma unroll
      for (int nt = 0; nt < 4; ++nt)
        store_c(XBl, PX, gm0 + mt * 16, gn0 + nt * 16, acc[mt][nt], false);
  }
  __syncthreads();

  // ---- out = z @ W3^T + b3  ([64,16]; waves 0..3, one 16x16 tile each) ------
  if (wave < 4) {
    const int m0f = wave * 16;
    v8f acc;
    const float bv = b3[r];
#pragma unroll
    for (int j = 0; j < 8; ++j) acc[j] = bv;
#pragma unroll
    for (int kt = 0; kt < 8; ++kt) {
      v16bf a  = load_a(XBl, PX, m0f, kt * 32);
      v16bf bw = load_b_packed(W3p, kt * 512);
      acc = wmma_bf16(a, bw, acc);
    }
    float* outb = out + (size_t)b * NNODE * ACTD;
#pragma unroll
    for (int j = 0; j < 8; ++j)
      outb[(m0f + g * 8 + j) * ACTD + r] = acc[j];
  }
}

// ---------------------------------------------------------------------------
// Prepass: pack fp32 weight W[O][D] into bf16 B-fragment order:
// out[((nt*KT+kt)*32 + lane)*16 + i] = W[nt*16 + (lane&15)][kt*32 + (lane>>4)*16 + i]
// ---------------------------------------------------------------------------
__global__ void pack_weight(const float* __restrict__ W, bf16* __restrict__ out,
                            int O, int D) {
  const int KT = D >> 5;
  const int total = O * D;
  int idx = blockIdx.x * 256 + threadIdx.x;
  if (idx >= total) return;
  const int i    = idx & 15;
  const int lane = (idx >> 4) & 31;
  const int blk  = idx >> 9;
  const int kt   = blk % KT;
  const int nt   = blk / KT;
  const int rr   = lane & 15, gg = lane >> 4;
  const int o    = nt * 16 + rr;
  const int k    = kt * 32 + gg * 16 + i;
  out[idx] = (bf16)W[(size_t)o * D + k];
}

// ---------------------------------------------------------------------------
extern "C" void kernel_launch(void* const* d_in, const int* in_sizes, int n_in,
                              void* d_out, int out_size, void* d_ws, size_t ws_size,
                              hipStream_t stream) {
  (void)in_sizes; (void)n_in; (void)out_size; (void)ws_size;
  const float* obs = (const float*)d_in[0];
  const float* adj = (const float*)d_in[1];
  const float* W1 = (const float*)d_in[2];  const float* b1 = (const float*)d_in[3];
  const float* W2 = (const float*)d_in[4];  const float* b2 = (const float*)d_in[5];
  const float* Wq = (const float*)d_in[6];  const float* bq = (const float*)d_in[7];
  const float* Wk = (const float*)d_in[8];  const float* bk = (const float*)d_in[9];
  const float* Wv = (const float*)d_in[10]; const float* bv = (const float*)d_in[11];
  const float* Wo = (const float*)d_in[12]; const float* bo = (const float*)d_in[13];
  const float* Wc = (const float*)d_in[14]; const float* bc = (const float*)d_in[15];
  const float* W3 = (const float*)d_in[16]; const float* b3 = (const float*)d_in[17];

  // Workspace layout (bf16 elements)
  bf16* ws = (bf16*)d_ws;
  bf16* W1p = ws;                       // 256*128
  bf16* W2p = W1p + 256 * 128;          // 256*256
  bf16* Wqp = W2p + 256 * 256;
  bf16* Wkp = Wqp + 256 * 256;
  bf16* Wvp = Wkp + 256 * 256;
  bf16* Wop = Wvp + 256 * 256;
  bf16* Wcp = Wop + 256 * 256;
  bf16* W3p = Wcp + 256 * 256;          // 16*256

  // Pack all weights into WMMA B-fragment order (runs every call; cheap).
  pack_weight<<<(256 * 128 + 255) / 256, 256, 0, stream>>>(W1, W1p, 256, 128);
  pack_weight<<<(256 * 256 + 255) / 256, 256, 0, stream>>>(W2, W2p, 256, 256);
  pack_weight<<<(256 * 256 + 255) / 256, 256, 0, stream>>>(Wq, Wqp, 256, 256);
  pack_weight<<<(256 * 256 + 255) / 256, 256, 0, stream>>>(Wk, Wkp, 256, 256);
  pack_weight<<<(256 * 256 + 255) / 256, 256, 0, stream>>>(Wv, Wvp, 256, 256);
  pack_weight<<<(256 * 256 + 255) / 256, 256, 0, stream>>>(Wo, Wop, 256, 256);
  pack_weight<<<(256 * 256 + 255) / 256, 256, 0, stream>>>(Wc, Wcp, 256, 256);
  pack_weight<<<(16  * 256 + 255) / 256, 256, 0, stream>>>(W3, W3p, 16, 256);

  // Allow >64KB dynamic LDS (CDNA5 WGP has 320KB).
  (void)hipFuncSetAttribute((const void*)sag_fused,
                            hipFuncAttributeMaxDynamicSharedMemorySize, SMEM_BYTES);

  sag_fused<<<BATCH, 256, SMEM_BYTES, stream>>>(
      obs, adj, W1p, b1, W2p, b2, Wqp, bq, Wkp, bk, Wvp, bv,
      Wop, bo, Wcp, bc, W3p, b3, (float*)d_out);
}
